// LinearCrossModalAttention_10814727652014
// MI455X (gfx1250) — matrix operand order repaired and checked by reading, not compiled, not run on hardware
//
#include <hip/hip_runtime.h>

// ---------------------------------------------------------------------------
// LinearCrossModalAttention for MI455X (gfx1250, wave32, WMMA bf16)
//   B=8, C=512, H=W=64, N=4096
//   q = Wq@xq ; k = Wk@xk ; v = Wv@xk
//   attn = softmax((q@k^T) * C^-0.5)
//   out  = query + Wo@(attn@v)
// Matmuls via v_wmma_f32_16x16x32_bf16 with double-buffered LDS staged by
// global_load_async_to_lds_b128 (ASYNCcnt pipeline) where layouts allow.
// ---------------------------------------------------------------------------

typedef __attribute__((ext_vector_type(16))) __bf16 v16bf;
typedef __attribute__((ext_vector_type(8)))  float  v8f;

#define BATCH 8
#define CDIM  512
#define NSPAT 4096

// ---- CDNA5 async staging helpers -------------------------------------------

// 16B per lane: memory -> LDS, tracked by ASYNCcnt (no VGPR data round-trip).
__device__ inline void async_copy_b128(unsigned lds_addr, const void* gaddr) {
    asm volatile("global_load_async_to_lds_b128 %0, %1, off"
                 :: "v"(lds_addr), "v"((unsigned long long)gaddr)
                 : "memory");
}
__device__ inline void wait_async0() {
    asm volatile("s_wait_asynccnt 0x0" ::: "memory");
}
// Generic shared-pointer -> LDS byte offset (low 32 bits of LDS aperture addr).
__device__ inline unsigned lds_off(const void* p) {
    return (unsigned)(unsigned long long)p;
}

// ---- fragment assembly from LDS (ISA 7.12.2 layouts) ----------------------

union Frag16 { uint4 u[2]; v16bf v; };

// A-matrix 16x32 bf16 fragment. p -> &As[mbase + (lane&15)][ (lane>>4)*8 ].
// elems 0..7  <- K = half*8 + 0..7      (p[0..7])
// elems 8..15 <- K = 16 + half*8 + 0..7 (p[16..23])
__device__ inline v16bf load_frag_a(const __bf16* p) {
    Frag16 f;
    f.u[0] = *reinterpret_cast<const uint4*>(p);
    f.u[1] = *reinterpret_cast<const uint4*>(p + 16);
    return f.v;
}

// B-matrix 32x16 bf16 fragment (K-contiguous storage).
// p -> &Bs[nbase + (lane&15)][ (lane>>4)*8 ] ; K = half*8+0..7, 16+half*8+0..7.
__device__ inline v16bf load_frag_b(const __bf16* p) {
    Frag16 f;
    f.u[0] = *reinterpret_cast<const uint4*>(p);
    f.u[1] = *reinterpret_cast<const uint4*>(p + 16);
    return f.v;
}

// ---- generic bf16 WMMA GEMM ------------------------------------------------
// C[M,Nout] = A[M,K] * B, where
//   TRANSB=0 : B is [K,Nout] row-major (ldb)
//   TRANSB=1 : B is [Nout,K] row-major (ldb)   (i.e. multiply by B^T)
//   B_F32    : B operand is f32 in global; converted to bf16 on LDS fill
//   OUT_BF16 : store bf16 result, else f32 (+ optional residual add)
// Block tile: 128x128, BK=32, 256 threads (8 waves, each 32x64).
// Double-buffered LDS; A (and B when TRANSB) staged via async-to-LDS.

template<bool TRANSB, bool B_F32, bool OUT_BF16, bool RESIDUAL>
__global__ __launch_bounds__(256)
void gemm_wmma_bf16(const __bf16* __restrict__ A, size_t sA,
                    const void*   __restrict__ Bv, size_t sB,
                    float*  __restrict__ outF,
                    __bf16* __restrict__ outB, size_t sO,
                    const float* __restrict__ Res, size_t sR,
                    int M, int Nout, int Kdim, int lda, int ldb)
{
    __shared__ __bf16 As[2][128][40];   // 80B row stride keeps 16B alignment
    __shared__ __bf16 Bs[2][128][40];   // stored [n][k] : K contiguous

    const int t  = threadIdx.x;
    const int bz = blockIdx.z;
    const int rowBase = blockIdx.y * 128;
    const int colBase = blockIdx.x * 128;

    A += (size_t)bz * sA;
    const __bf16* Bh = reinterpret_cast<const __bf16*>(Bv) + (size_t)bz * sB;
    const float*  Bf = reinterpret_cast<const float*>(Bv)  + (size_t)bz * sB;

    const int lane = t & 31;
    const int wid  = t >> 5;
    const int wm   = wid >> 1;     // 0..3 -> 32-row strip
    const int wn   = wid & 1;      // 0..1 -> 64-col strip
    const int r16  = lane & 15;
    const int half = lane >> 4;

    // staging thread mappings
    const int tr = t >> 2;          // 0..63   (contiguous b128 fills)
    const int tc = (t & 3) << 3;    // 0,8,16,24
    const int kk = t >> 3;          // 0..31   (transpose fill, non-trans B)
    const int nn = (t & 7) << 4;    // 0..112

    v8f acc[2][4];
#pragma unroll
    for (int i = 0; i < 2; ++i)
#pragma unroll
        for (int j = 0; j < 4; ++j) acc[i][j] = (v8f){0.f,0.f,0.f,0.f,0.f,0.f,0.f,0.f};

    uint4 breg[4];   // in-flight B tile (non-trans path)

    // ---- tile staging -------------------------------------------------------
    auto issue_tile = [&](int k0, int b) {
        // A tile: rows rowBase..+127, cols k0..+31 -> async b128 per lane x2
#pragma unroll
        for (int p = 0; p < 2; ++p) {
            const int rr = tr + p * 64;
            async_copy_b128(lds_off(&As[b][rr][tc]),
                            A + (size_t)(rowBase + rr) * lda + k0 + tc);
        }
        if (TRANSB) {
#pragma unroll
            for (int p = 0; p < 2; ++p) {
                const int rr = tr + p * 64;
                async_copy_b128(lds_off(&Bs[b][rr][tc]),
                                Bh + (size_t)(colBase + rr) * ldb + k0 + tc);
            }
        } else if (B_F32) {
            const uint4* gp = reinterpret_cast<const uint4*>(
                Bf + (size_t)(k0 + kk) * ldb + colBase + nn);
#pragma unroll
            for (int i = 0; i < 4; ++i) breg[i] = gp[i];
        } else {
            const uint4* gp = reinterpret_cast<const uint4*>(
                Bh + (size_t)(k0 + kk) * ldb + colBase + nn);
            breg[0] = gp[0];
            breg[1] = gp[1];
        }
    };
    auto commit_tile = [&](int b) {   // VGPR -> LDS transpose (non-trans B only)
        if (!TRANSB) {
            if (B_F32) {
                const float* f = reinterpret_cast<const float*>(breg);
#pragma unroll
                for (int i = 0; i < 16; ++i) Bs[b][nn + i][kk] = (__bf16)f[i];
            } else {
                const __bf16* h = reinterpret_cast<const __bf16*>(breg);
#pragma unroll
                for (int i = 0; i < 16; ++i) Bs[b][nn + i][kk] = h[i];
            }
        }
    };

    const int nT = Kdim >> 5;

    // prologue: tile 0 -> buffer 0
    issue_tile(0, 0);
    commit_tile(0);
    wait_async0();
    __syncthreads();

    for (int kt = 0; kt < nT; ++kt) {
        const int  cur  = kt & 1;
        const int  nxt  = cur ^ 1;
        const bool more = (kt + 1) < nT;

        if (more) issue_tile((kt + 1) << 5, nxt);   // overlaps WMMA below

        // ---- 2x4 grid of 16x16x32 WMMAs from buffer `cur`
        v16bf af[2], bfg[4];
#pragma unroll
        for (int mf = 0; mf < 2; ++mf)
            af[mf] = load_frag_a(&As[cur][wm * 32 + mf * 16 + r16][half * 8]);
#pragma unroll
        for (int nf = 0; nf < 4; ++nf)
            bfg[nf] = load_frag_b(&Bs[cur][wn * 64 + nf * 16 + r16][half * 8]);

#pragma unroll
        for (int mf = 0; mf < 2; ++mf)
#pragma unroll
            for (int nf = 0; nf < 4; ++nf)
                acc[mf][nf] = __builtin_amdgcn_wmma_f32_16x16x32_bf16(
                    false, af[mf], false, bfg[nf], (short)0, acc[mf][nf],
                    false, false);

        if (more) {
            commit_tile(nxt);   // LDS transpose of prefetched B (non-trans)
            wait_async0();      // drain this wave's async writes for tile kt+1
        }
        __syncthreads();
    }

    // ---- epilogue: C/D layout: VGPR v -> M = v + 8*half ; N = lane&15
    const int mo = half * 8;
    outB += (size_t)bz * sO;
    outF += (size_t)bz * sO;
    Res  += (size_t)bz * sR;
#pragma unroll
    for (int mf = 0; mf < 2; ++mf) {
#pragma unroll
        for (int nf = 0; nf < 4; ++nf) {
            const int gcol = colBase + wn * 64 + nf * 16 + r16;
#pragma unroll
            for (int v = 0; v < 8; ++v) {
                const int grow = rowBase + wm * 32 + mf * 16 + mo + v;
                const size_t idx = (size_t)grow * Nout + gcol;
                const float val = acc[mf][nf][v];
                if (OUT_BF16) {
                    outB[idx] = (__bf16)val;
                } else {
                    float rr = val;
                    if (RESIDUAL) rr += Res[idx];
                    outF[idx] = rr;
                }
            }
        }
    }
}

// ---- f32 -> bf16 conversion (weights) --------------------------------------
__global__ __launch_bounds__(256)
void cvt_f32_to_bf16(const float* __restrict__ in, __bf16* __restrict__ out, int n)
{
    const int i = blockIdx.x * 256 + threadIdx.x;
    if (i < n) out[i] = (__bf16)in[i];
}

// ---- scaled row softmax: S[row,0..511]*scale -> bf16 attn ------------------
__global__ __launch_bounds__(256)
void softmax512(const float* __restrict__ S, __bf16* __restrict__ Attn, float scale)
{
    __shared__ float red[256];
    const int row = blockIdx.x;
    const int t = threadIdx.x;
    const float* src = S + (size_t)row * 512;

    float a = src[t] * scale;
    float b = src[t + 256] * scale;

    red[t] = fmaxf(a, b);
    __syncthreads();
#pragma unroll
    for (int s = 128; s > 0; s >>= 1) {
        if (t < s) red[t] = fmaxf(red[t], red[t + s]);
        __syncthreads();
    }
    const float rowmax = red[0];
    __syncthreads();

    const float ea = __expf(a - rowmax);
    const float eb = __expf(b - rowmax);
    red[t] = ea + eb;
    __syncthreads();
#pragma unroll
    for (int s = 128; s > 0; s >>= 1) {
        if (t < s) red[t] += red[t + s];
        __syncthreads();
    }
    const float inv = 1.0f / red[0];

    __bf16* dst = Attn + (size_t)row * 512;
    dst[t]       = (__bf16)(ea * inv);
    dst[t + 256] = (__bf16)(eb * inv);
}

// ---------------------------------------------------------------------------
extern "C" void kernel_launch(void* const* d_in, const int* in_sizes, int n_in,
                              void* d_out, int out_size, void* d_ws, size_t ws_size,
                              hipStream_t stream)
{
    (void)in_sizes; (void)n_in; (void)out_size; (void)ws_size;

    const float* query = (const float*)d_in[0];   // [8,512,64,64]
    const float* keyf  = (const float*)d_in[1];   // [8,512,64,64]
    const float* Wq    = (const float*)d_in[2];   // [512,512]
    const float* Wk    = (const float*)d_in[3];
    const float* Wv    = (const float*)d_in[4];
    const float* Wo    = (const float*)d_in[5];

    constexpr size_t SZ_W   = (size_t)CDIM * CDIM * 2;            // 0.5 MB bf16
    constexpr size_t SZ_T   = (size_t)BATCH * CDIM * NSPAT * 2;   // 32 MB bf16
    constexpr size_t SZ_S   = (size_t)BATCH * CDIM * CDIM * 4;    // 8 MB f32
    constexpr size_t OFF_WQ = 0;
    constexpr size_t OFF_WK = OFF_WQ + SZ_W;
    constexpr size_t OFF_WV = OFF_WK + SZ_W;
    constexpr size_t OFF_WO = OFF_WV + SZ_W;
    constexpr size_t OFF_Q  = OFF_WO + SZ_W;
    constexpr size_t OFF_K  = OFF_Q + SZ_T;
    constexpr size_t OFF_V  = OFF_K + SZ_T;
    constexpr size_t OFF_S  = OFF_V + SZ_T;
    constexpr size_t OFF_A  = OFF_S + SZ_S;
    // O (attn@V result) reuses the Q region: Q is dead once S is computed.

    char* ws = (char*)d_ws;
    __bf16* wq_b = (__bf16*)(ws + OFF_WQ);
    __bf16* wk_b = (__bf16*)(ws + OFF_WK);
    __bf16* wv_b = (__bf16*)(ws + OFF_WV);
    __bf16* wo_b = (__bf16*)(ws + OFF_WO);
    __bf16* Qb   = (__bf16*)(ws + OFF_Q);
    __bf16* Kb   = (__bf16*)(ws + OFF_K);
    __bf16* Vb   = (__bf16*)(ws + OFF_V);
    float*  Sb   = (float*) (ws + OFF_S);
    __bf16* Ab   = (__bf16*)(ws + OFF_A);
    __bf16* Ob   = Qb;

    const size_t strT = (size_t)CDIM * NSPAT;   // per-batch activation stride
    const size_t strS = (size_t)CDIM * CDIM;    // per-batch S/attn stride

    // 1) weights -> bf16
    {
        dim3 g((CDIM * CDIM + 255) / 256);
        cvt_f32_to_bf16<<<g, 256, 0, stream>>>(Wq, wq_b, CDIM * CDIM);
        cvt_f32_to_bf16<<<g, 256, 0, stream>>>(Wk, wk_b, CDIM * CDIM);
        cvt_f32_to_bf16<<<g, 256, 0, stream>>>(Wv, wv_b, CDIM * CDIM);
        cvt_f32_to_bf16<<<g, 256, 0, stream>>>(Wo, wo_b, CDIM * CDIM);
    }

    // 2) projections: Q = Wq@xq ; K = Wk@xk ; V = Wv@xk   (B operand f32->bf16)
    {
        dim3 g(NSPAT / 128, CDIM / 128, BATCH);
        gemm_wmma_bf16<false, true, true, false><<<g, 256, 0, stream>>>(
            wq_b, 0, query, strT, nullptr, Qb, strT, nullptr, 0,
            CDIM, NSPAT, CDIM, CDIM, NSPAT);
        gemm_wmma_bf16<false, true, true, false><<<g, 256, 0, stream>>>(
            wk_b, 0, keyf, strT, nullptr, Kb, strT, nullptr, 0,
            CDIM, NSPAT, CDIM, CDIM, NSPAT);
        gemm_wmma_bf16<false, true, true, false><<<g, 256, 0, stream>>>(
            wv_b, 0, keyf, strT, nullptr, Vb, strT, nullptr, 0,
            CDIM, NSPAT, CDIM, CDIM, NSPAT);
    }

    // 3) S = Q @ K^T   (f32 output; both operands async-staged; scale in softmax)
    {
        dim3 g(CDIM / 128, CDIM / 128, BATCH);
        gemm_wmma_bf16<true, false, false, false><<<g, 256, 0, stream>>>(
            Qb, strT, Kb, strT, Sb, nullptr, strS, nullptr, 0,
            CDIM, CDIM, NSPAT, NSPAT, NSPAT);
    }

    // 4) attn = softmax(S * C^-0.5) -> bf16
    {
        const float scale = 0.044194173824159216f;  // 512^-0.5
        softmax512<<<BATCH * CDIM, 256, 0, stream>>>(Sb, Ab, scale);
    }

    // 5) O = attn @ V   (bf16, reuses Q region)
    {
        dim3 g(NSPAT / 128, CDIM / 128, BATCH);
        gemm_wmma_bf16<false, false, true, false><<<g, 256, 0, stream>>>(
            Ab, strS, Vb, strT, nullptr, Ob, strT, nullptr, 0,
            CDIM, NSPAT, CDIM, CDIM, NSPAT);
    }

    // 6) out = query + Wo @ O   (f32, residual fused)
    {
        dim3 g(NSPAT / 128, CDIM / 128, BATCH);
        gemm_wmma_bf16<false, false, false, true><<<g, 256, 0, stream>>>(
            wo_b, 0, Ob, strT, (float*)d_out, nullptr, strT, query, strT,
            CDIM, NSPAT, CDIM, CDIM, NSPAT);
    }
}
